// LoopedLatentController_36163624632391
// MI455X (gfx1250) — matrix operand order repaired and checked by reading, not compiled, not run on hardware
//
#include <hip/hip_runtime.h>
#include <hip/hip_bf16.h>
#include <math.h>

// ---------------------------------------------------------------------------
// LoopedLatentController forward on MI455X (gfx1250, wave32, WMMA).
// Compute-bound -> everything through v_wmma_f32_16x16x32_bf16. LDS layouts
// give two ds_load_b128 per fragment; GEMM A tiles stream via the Tensor Data
// Mover (tensor_load_to_lds, TENSORcnt), falling back to async b128 copies.
// ---------------------------------------------------------------------------

typedef __bf16 bf16_t;
typedef __attribute__((ext_vector_type(16))) __bf16        v16bf;
typedef __attribute__((ext_vector_type(8)))  float         v8f;
typedef __attribute__((ext_vector_type(4)))  unsigned int  uint4v;
typedef __attribute__((ext_vector_type(2)))  unsigned int  uint2v;
typedef __attribute__((ext_vector_type(4)))  float         float4v;

#define B_       2
#define T_       1024
#define V_       32000
#define D_       1024
#define H_       16
#define HD_      64
#define L_       8
#define FF_      4096
#define N_SLOTS_ 64
#define N_MEM_   66
#define S_       (T_ + N_MEM_)   // 1090
#define EPS_     1e-8f
#define NEG_INF_ (-3.0e38f)

// ---- gfx1250 data-movement feature detection ------------------------------
#if defined(__HIP_DEVICE_COMPILE__) &&                                   \
    __has_builtin(__builtin_amdgcn_global_load_async_to_lds_b128) &&     \
    __has_builtin(__builtin_amdgcn_s_wait_asynccnt)
#define HAVE_ASYNC 1
typedef int gcc_int4_t __attribute__((vector_size(16)));
typedef __attribute__((address_space(1))) gcc_int4_t* gptr_b128_t;
typedef __attribute__((address_space(3))) gcc_int4_t* lptr_b128_t;
#else
#define HAVE_ASYNC 0
#endif

#if defined(__HIP_DEVICE_COMPILE__) &&                                   \
    __has_builtin(__builtin_amdgcn_tensor_load_to_lds) &&                \
    __has_builtin(__builtin_amdgcn_s_wait_tensorcnt)
#define HAVE_TDM 1
typedef __attribute__((ext_vector_type(4))) unsigned int u32x4_t;
typedef __attribute__((ext_vector_type(8))) int          i32x8_t;
typedef __attribute__((ext_vector_type(4))) int          i32x4_t;
typedef __attribute__((address_space(3))) void           tdm_lvoid_t;
#else
#define HAVE_TDM 0
#endif

#if HAVE_TDM
// TDM 2D tile load: rows x cols (bf16, 2B elements), row stride in elements.
// D# per CDNA5 ISA ch.8: group0 = {count/type, lds_addr, global_addr},
// group1 = {data_size, tensor dims (OOB zero-fill), tile dims, dim0 stride}.
// This toolchain exposes the 6-arg builtin: (g0, g1, g2, g3, g4, cpol).
__device__ __forceinline__ void tdm_load_2d(const bf16_t* g, unsigned lds_off,
                                            int rows_avail, int cols_avail,
                                            int row_stride_elems,
                                            int tile_rows, int tile_cols) {
  const unsigned long long ga = (unsigned long long)(size_t)g;
  u32x4_t g0;
  g0[0] = 1u;                                                   // count=1, user
  g0[1] = lds_off;                                              // bits 63:32
  g0[2] = (unsigned)ga;                                         // addr[31:0]
  g0[3] = (unsigned)((ga >> 32) & 0x01ffffffu) | 0x80000000u;   // addr[56:32] | type=2
  i32x8_t g1;
  g1[0] = 1 << 16;                                              // data_size = 2B
  g1[1] = (int)(((unsigned)cols_avail & 0xffffu) << 16);        // td0[15:0]
  g1[2] = (int)(((unsigned)cols_avail >> 16) |
                (((unsigned)rows_avail & 0xffffu) << 16));      // td0[31:16]|td1[15:0]
  g1[3] = (int)(((unsigned)rows_avail >> 16) |
                ((unsigned)tile_cols << 16));                   // td1[31:16]|tile0
  g1[4] = tile_rows;                                            // tile1 (tile2=0)
  g1[5] = row_stride_elems;                                     // dim0 stride lo
  g1[6] = 0;
  g1[7] = 0;
  const i32x4_t gz4 = {0, 0, 0, 0};
  const i32x8_t gz8 = {0, 0, 0, 0, 0, 0, 0, 0};
  __builtin_amdgcn_tensor_load_to_lds(g0, g1, gz4, gz4, gz8, 0);
}
#endif

// ---------------------------------------------------------------------------
// WMMA helpers
// ---------------------------------------------------------------------------
__device__ __forceinline__ v8f wmma_bf16(v16bf a, v16bf b, v8f c) {
  return __builtin_amdgcn_wmma_f32_16x16x32_bf16(
      false, a, false, b, (short)0, c, false, false);
}

// Fragment loader: 8 contiguous bf16 at (row, off0) and 8 at (row, off1),
// both 16B-aligned -> two ds_load_b128.
__device__ __forceinline__ v16bf load_frag(const bf16_t* lds, int row, int ld,
                                           int off0, int off1) {
  union { v16bf v; uint4v q[2]; } f;
  const bf16_t* base = lds + row * ld;
  f.q[0] = *(const uint4v*)(base + off0);
  f.q[1] = *(const uint4v*)(base + off1);
  return f.v;
}

// ---------------------------------------------------------------------------
// Kernel 1: build x = [mem_start, memory, mem_end, embed[tokens]]
// ---------------------------------------------------------------------------
__global__ __launch_bounds__(256) void build_x_kernel(
    const int* __restrict__ tok, const float* __restrict__ mem,
    const float* __restrict__ embed, float* __restrict__ x) {
  const int s = blockIdx.x, b = blockIdx.y, tid = threadIdx.x;
  const float* src;
  if (s == 0)                 src = embed + (size_t)1 * D_;
  else if (s <= N_SLOTS_)     src = mem + ((size_t)b * N_SLOTS_ + (s - 1)) * D_;
  else if (s == N_SLOTS_ + 1) src = embed + (size_t)2 * D_;
  else                        src = embed + (size_t)tok[b * T_ + (s - N_MEM_)] * D_;
  const float4v* s4 = (const float4v*)src;
  float4v* d4 = (float4v*)(x + ((size_t)b * S_ + s) * D_);
  d4[tid] = s4[tid];
}

// ---------------------------------------------------------------------------
// Kernel 2: RMSNorm -> bf16. grid (B*S), block 256
// ---------------------------------------------------------------------------
__global__ __launch_bounds__(256) void rmsnorm_kernel(
    const float* __restrict__ x, const float* __restrict__ w,
    bf16_t* __restrict__ out) {
  __shared__ float red[256];
  const int row = blockIdx.x, tid = threadIdx.x;
  const float4v xv = *(const float4v*)(x + (size_t)row * D_ + tid * 4);
  red[tid] = xv[0]*xv[0] + xv[1]*xv[1] + xv[2]*xv[2] + xv[3]*xv[3];
  __syncthreads();
  for (int s = 128; s > 0; s >>= 1) {
    if (tid < s) red[tid] += red[tid + s];
    __syncthreads();
  }
  const float n = rsqrtf(red[0] * (1.f / (float)D_) + EPS_);
  const float4v wv = *(const float4v*)(w + tid * 4);
  union { uint2v u; bf16_t e[4]; } o;
#pragma unroll
  for (int t = 0; t < 4; ++t) o.e[t] = (bf16_t)(xv[t] * n * wv[t]);
  *(uint2v*)(out + (size_t)row * D_ + tid * 4) = o.u;
}

// ---------------------------------------------------------------------------
// Kernel 3: tiled WMMA GEMM  C[M,N] = A[M,K](bf16) * W(fp32->bf16)
//   WBT==0: W is KxN row-major   WBT==1: W is NxK row-major (Bᵀ)
//   EPI: 0 fp32, 1 bf16, 2 fp32+residual, 3 SiLU->bf16
// block 256 (8 waves), macro-tile 128x64, BK=64 (8 WMMAs per barrier),
// ping-pong LDS double buffer, TDM (or async b128) staging for the A tile.
// ---------------------------------------------------------------------------
template <int EPI, int WBT>
__global__ __launch_bounds__(256) void gemm_wmma_kernel(
    const bf16_t* __restrict__ A, const float* __restrict__ W,
    float* __restrict__ Cf, bf16_t* __restrict__ Cb,
    const float* __restrict__ resid, int M, int N, int K) {
  __shared__ __align__(16) bf16_t sA[2][128 * 64];   // [m][k]
  __shared__ __align__(16) bf16_t sBt[2][64 * 64];   // [n][k] (transposed)
  const int tid  = threadIdx.x;
  const int lane = tid & 31;
  const int wave = tid >> 5;
  const int blockN = blockIdx.x * 64;
  const int blockM = blockIdx.y * 128;
  const int nk = K >> 6;            // K is a multiple of 64 (1024/4096)

  v8f acc[4];
#pragma unroll
  for (int nt = 0; nt < 4; ++nt)
#pragma unroll
    for (int i = 0; i < 8; ++i) acc[nt][i] = 0.f;

  // B-tile load roles (64 n x 64 k fp32, 16 values per thread)
  const int bn = WBT ? (tid >> 2) : (tid & 63);
  const int bk = WBT ? ((tid & 3) * 16) : ((tid >> 6) * 16);
  float bR[16];
#if !HAVE_TDM && !HAVE_ASYNC
  uint4v aR[4];
#endif

  auto stageA = [&](int k0, int buf) {
#if HAVE_TDM
    if (wave == 0) {   // one TDM descriptor moves the whole 128x64 tile
      const bf16_t* g = A + (size_t)blockM * K + k0;
      const unsigned lds_off = (unsigned)(size_t)(tdm_lvoid_t*)(&sA[buf][0]);
      tdm_load_2d(g, lds_off, M - blockM, K - k0, K, 128, 64);
    }
#else
#pragma unroll
    for (int j = 0; j < 4; ++j) {
      const int u = tid + j * 256;       // 1024 uint4 chunks: row=u>>3, c=(u&7)*8
      int gm = blockM + (u >> 3);
      if (gm > M - 1) gm = M - 1;        // clamp: row-local garbage is store-guarded
      const bf16_t* g = A + (size_t)gm * K + k0 + (u & 7) * 8;
#if HAVE_ASYNC
      __builtin_amdgcn_global_load_async_to_lds_b128(
          (gptr_b128_t)g, (lptr_b128_t)(&sA[buf][u * 8]), 0, 0);
#else
      aR[j] = *(const uint4v*)g;
#endif
    }
#if !HAVE_ASYNC
#pragma unroll
    for (int j = 0; j < 4; ++j) {
      const int u = tid + j * 256;
      *(uint4v*)(&sA[buf][u * 8]) = aR[j];
    }
#endif
#endif
  };

  auto waitStage = [&]() {
#if HAVE_TDM
    if (wave == 0) __builtin_amdgcn_s_wait_tensorcnt(0);
#elif HAVE_ASYNC
    __builtin_amdgcn_s_wait_asynccnt(0);
#endif
  };

  auto loadB_regs = [&](int k0) {
    if (WBT) {  // W[n][k]: 4 contiguous float4 per thread
      const float* g = W + (size_t)(blockN + bn) * K + k0 + bk;
#pragma unroll
      for (int j = 0; j < 4; ++j) {
        const float4v xq = *(const float4v*)(g + j * 4);
#pragma unroll
        for (int t = 0; t < 4; ++t) bR[j * 4 + t] = xq[t];
      }
    } else {    // W[k][n]: 16 k-strided loads (coalesced across lanes in n)
      const float* g = W + (size_t)(k0 + bk) * N + (blockN + bn);
#pragma unroll
      for (int t = 0; t < 16; ++t) bR[t] = g[(size_t)t * N];
    }
  };

  auto storeB = [&](int buf) {
#pragma unroll
    for (int j = 0; j < 2; ++j) {
      union { uint4v q; bf16_t e[8]; } p;
#pragma unroll
      for (int t = 0; t < 8; ++t) p.e[t] = (bf16_t)bR[j * 8 + t];
      *(uint4v*)(&sBt[buf][bn * 64 + bk + j * 8]) = p.q;
    }
  };

  // Prologue: tile 0
  stageA(0, 0);
  loadB_regs(0);
  storeB(0);
  waitStage();
  __syncthreads();

  const int grp  = (lane >> 4) * 8;
  const int kb   = (lane >> 4) * 16;
  const int mrow = wave * 16 + (lane & 15);
  const int ncol = lane & 15;

  for (int kt = 0; kt < nk; ++kt) {
    const int cur = kt & 1, nxt = cur ^ 1;
    const bool more = (kt + 1) < nk;
    if (more) {
      stageA((kt + 1) * 64, nxt);      // TDM/async copies in flight during WMMAs
      loadB_regs((kt + 1) * 64);
      if (kt + 2 < nk) {               // L2 prefetch of tile k+2 weights
        const int k2 = (kt + 2) * 64;
        const float* pf = WBT ? W + (size_t)(blockN + bn) * K + k2 + bk
                              : W + (size_t)(k2 + bk) * N + (blockN + bn);
        __builtin_prefetch(pf, 0, 1);
      }
    }
    const v16bf af0 = load_frag(&sA[cur][0], mrow, 64, grp,      grp + 16);
    const v16bf af1 = load_frag(&sA[cur][0], mrow, 64, 32 + grp, 48 + grp);
#pragma unroll
    for (int nt = 0; nt < 4; ++nt) {
      const v16bf b0 = load_frag(&sBt[cur][0], nt * 16 + ncol, 64, grp,      grp + 16);
      const v16bf b1 = load_frag(&sBt[cur][0], nt * 16 + ncol, 64, 32 + grp, 48 + grp);
      acc[nt] = wmma_bf16(af0, b0, acc[nt]);
      acc[nt] = wmma_bf16(af1, b1, acc[nt]);
    }
    if (more) storeB(nxt);
    waitStage();
    __syncthreads();
  }

  // Epilogue. C layout: element i -> row (lane>>4)*8+i, col = lane&15.
  const int rbase = (lane >> 4) * 8;
#pragma unroll
  for (int nt = 0; nt < 4; ++nt) {
    const int gn = blockN + nt * 16 + ncol;
    if (gn >= N) continue;
#pragma unroll
    for (int i = 0; i < 8; ++i) {
      const int gm = blockM + wave * 16 + rbase + i;
      if (gm >= M) continue;
      const size_t off = (size_t)gm * N + gn;
      const float v = acc[nt][i];
      if (EPI == 0)      Cf[off] = v;
      else if (EPI == 1) Cb[off] = (bf16_t)v;
      else if (EPI == 2) Cf[off] = resid[off] + v;
      else               Cb[off] = (bf16_t)(v / (1.f + __expf(-v)));  // SiLU
    }
  }
}

// ---------------------------------------------------------------------------
// Kernel 4: RoPE on q,k (bf16, B,S,H,HD). grid (S, H, B), block 32
// ---------------------------------------------------------------------------
__global__ __launch_bounds__(32) void rope_kernel(bf16_t* __restrict__ q,
                                                  bf16_t* __restrict__ k) {
  const int s = blockIdx.x, h = blockIdx.y, b = blockIdx.z, d = threadIdx.x;
  const float freq = __powf(10000.f, -(float)d * (1.f / 32.f));
  float sn, c;
  __sincosf((float)s * freq, &sn, &c);
  const size_t base = (((size_t)b * S_ + s) * H_ + h) * (size_t)HD_;
  float x1 = (float)q[base + d], x2 = (float)q[base + 32 + d];
  q[base + d]      = (bf16_t)(x1 * c - x2 * sn);
  q[base + 32 + d] = (bf16_t)(x1 * sn + x2 * c);
  x1 = (float)k[base + d]; x2 = (float)k[base + 32 + d];
  k[base + d]      = (bf16_t)(x1 * c - x2 * sn);
  k[base + 32 + d] = (bf16_t)(x1 * sn + x2 * c);
}

// ---------------------------------------------------------------------------
// Kernel 5: flash attention. 4 waves/block, each owns one 16-query tile and a
// private LDS slice. Scores (Q·Kᵀ) and PV via WMMA; online softmax with
// width-16 shfl_xor; V staged transposed [d][key] for contiguous PV B-frags.
// ---------------------------------------------------------------------------
__global__ __launch_bounds__(128) void attention_kernel(
    const bf16_t* __restrict__ Q, const bf16_t* __restrict__ K,
    const bf16_t* __restrict__ Vv, bf16_t* __restrict__ O) {
  __shared__ __align__(16) bf16_t sQ [4][16 * 64];
  __shared__ __align__(16) bf16_t sK [4][32 * 64];   // [key][d]
  __shared__ __align__(16) bf16_t sVt[4][64 * 32];   // [d][key]
  __shared__ __align__(16) bf16_t sP [4][16 * 32];   // [row][key]
  const int tid  = threadIdx.x;
  const int lane = tid & 31;
  const int wave = tid >> 5;
  int t = blockIdx.x * 4 + wave;
  if (t > (S_ + 15) / 16 - 1) t = (S_ + 15) / 16 - 1;  // duplicate tail work
  const int q0 = t * 16;
  const int h = blockIdx.y;
  const int b = blockIdx.z;
  const float scale = 0.125f;  // 1/sqrt(64)
  bf16_t* sQw = sQ[wave];
  bf16_t* sKw = sK[wave];
  bf16_t* sVw = sVt[wave];
  bf16_t* sPw = sP[wave];

#pragma unroll
  for (int j = 0; j < 4; ++j) {
    const int u = lane + 32 * j;
    const int r = u >> 3, c = (u & 7) * 8;
    const int qi = (q0 + r < S_) ? q0 + r : S_ - 1;
    *(uint4v*)(&sQw[u * 8]) =
        *(const uint4v*)(Q + (((size_t)b * S_ + qi) * H_ + h) * HD_ + c);
  }
  __syncthreads();
  const int grp = (lane >> 4) * 8;
  const int kb  = (lane >> 4) * 16;
  const int col = lane & 15;
  const int half = lane >> 4;
  const v16bf aq0 = load_frag(sQw, col, 64, grp, grp + 16);
  const v16bf aq1 = load_frag(sQw, col, 64, 32 + grp, 48 + grp);

  v8f accO[4];
  float m_run[8], l_run[8];
#pragma unroll
  for (int nt = 0; nt < 4; ++nt)
#pragma unroll
    for (int i = 0; i < 8; ++i) accO[nt][i] = 0.f;
#pragma unroll
  for (int i = 0; i < 8; ++i) { m_run[i] = NEG_INF_; l_run[i] = 0.f; }

  for (int c0 = 0; c0 < S_; c0 += 32) {
    __syncthreads();
#pragma unroll
    for (int j = 0; j < 8; ++j) {
      const int u = lane + 32 * j;
      const int r = u >> 3, c = (u & 7) * 8;
      const int kj = c0 + r;
      union { uint4v q; bf16_t e[8]; } kvv, vvv;
      if (kj < S_) {
        const size_t g = (((size_t)b * S_ + kj) * H_ + h) * HD_ + c;
        kvv.q = *(const uint4v*)(K + g);
        vvv.q = *(const uint4v*)(Vv + g);
      } else {
        kvv.q = (uint4v){0u, 0u, 0u, 0u};
        vvv.q = kvv.q;
      }
      *(uint4v*)(&sKw[u * 8]) = kvv.q;
#pragma unroll
      for (int e = 0; e < 8; ++e) sVw[(c + e) * 32 + r] = vvv.e[e];
    }
    __syncthreads();

    v8f s0 = {0, 0, 0, 0, 0, 0, 0, 0};
    v8f s1 = {0, 0, 0, 0, 0, 0, 0, 0};
    s0 = wmma_bf16(aq0, load_frag(sKw, col,      64, grp,      grp + 16), s0);
    s0 = wmma_bf16(aq1, load_frag(sKw, col,      64, 32 + grp, 48 + grp), s0);
    s1 = wmma_bf16(aq0, load_frag(sKw, 16 + col, 64, grp,      grp + 16), s1);
    s1 = wmma_bf16(aq1, load_frag(sKw, 16 + col, 64, 32 + grp, 48 + grp), s1);

    float fac[8];
#pragma unroll
    for (int i = 0; i < 8; ++i) {
      const int r = half * 8 + i;
      const int qi = q0 + r;
      const int k0i = c0 + col, k1i = c0 + 16 + col;
      const bool vis0 = (k0i < S_) && ((k0i < N_MEM_) || (qi >= N_MEM_ && k0i <= qi));
      const bool vis1 = (k1i < S_) && ((k1i < N_MEM_) || (qi >= N_MEM_ && k1i <= qi));
      const float e0 = vis0 ? s0[i] * scale : NEG_INF_;
      const float e1 = vis1 ? s1[i] * scale : NEG_INF_;
      float mx = fmaxf(e0, e1);
#pragma unroll
      for (int off = 1; off < 16; off <<= 1) mx = fmaxf(mx, __shfl_xor(mx, off, 16));
      const float mnew = fmaxf(m_run[i], mx);
      const float f  = __expf(m_run[i] - mnew);
      const float p0 = __expf(e0 - mnew);
      const float p1 = __expf(e1 - mnew);
      float rs = p0 + p1;
#pragma unroll
      for (int off = 1; off < 16; off <<= 1) rs += __shfl_xor(rs, off, 16);
      l_run[i] = l_run[i] * f + rs;
      m_run[i] = mnew;
      fac[i] = f;
      sPw[r * 32 + col]      = (bf16_t)p0;
      sPw[r * 32 + 16 + col] = (bf16_t)p1;
    }
    __syncthreads();
#pragma unroll
    for (int nt = 0; nt < 4; ++nt)
#pragma unroll
      for (int i = 0; i < 8; ++i) accO[nt][i] *= fac[i];

    const v16bf ap = load_frag(sPw, col, 32, grp, grp + 16);
#pragma unroll
    for (int nt = 0; nt < 4; ++nt)
      accO[nt] = wmma_bf16(ap, load_frag(sVw, nt * 16 + col, 32, grp, grp + 16),
                           accO[nt]);
  }

#pragma unroll
  for (int i = 0; i < 8; ++i) {
    const int qi = q0 + half * 8 + i;
    if (qi >= S_) continue;
    const float inv = (l_run[i] > 0.f) ? 1.f / l_run[i] : 0.f;
#pragma unroll
    for (int nt = 0; nt < 4; ++nt)
      O[(((size_t)b * S_ + qi) * H_ + h) * HD_ + nt * 16 + col] =
          (bf16_t)(accO[nt][i] * inv);
  }
}

// ---------------------------------------------------------------------------
// Kernel 6: halt logits. grid (2, B), block 256
// ---------------------------------------------------------------------------
__global__ __launch_bounds__(256) void halt_kernel(
    const bf16_t* __restrict__ xn, const float* __restrict__ hW,
    const float* __restrict__ hb, float* __restrict__ out) {
  __shared__ float red[256];
  const int j = blockIdx.x, b = blockIdx.y, tid = threadIdx.x;
  const bf16_t* xr = xn + ((size_t)b * S_ + (S_ - 1)) * D_;
  const float* wr = hW + (size_t)j * D_;
  float s = 0.f;
  for (int i = tid; i < D_; i += 256) s += (float)xr[i] * wr[i];
  red[tid] = s;
  __syncthreads();
  for (int st = 128; st > 0; st >>= 1) {
    if (tid < st) red[tid] += red[tid + st];
    __syncthreads();
  }
  if (tid == 0) out[b * 2 + j] = red[0] + hb[j];
}

// ---------------------------------------------------------------------------
// Launcher
// ---------------------------------------------------------------------------
extern "C" void kernel_launch(void* const* d_in, const int* in_sizes, int n_in,
                              void* d_out, int out_size, void* d_ws,
                              size_t ws_size, hipStream_t stream) {
  (void)in_sizes; (void)n_in; (void)out_size; (void)ws_size;
  const int*   tok   = (const int*)d_in[0];
  const float* mem   = (const float*)d_in[1];
  const float* embed = (const float*)d_in[2];
  const float* Wq    = (const float*)d_in[3];
  const float* Wk    = (const float*)d_in[4];
  const float* Wv    = (const float*)d_in[5];
  const float* Wo    = (const float*)d_in[6];
  const float* n1    = (const float*)d_in[7];
  const float* n2    = (const float*)d_in[8];
  const float* Wup   = (const float*)d_in[9];
  const float* Wdn   = (const float*)d_in[10];
  const float* fnw   = (const float*)d_in[11];
  const float* hW    = (const float*)d_in[12];
  const float* hb    = (const float*)d_in[13];

  float* logits = (float*)d_out;
  float* halt   = logits + (size_t)B_ * T_ * V_;

  char* ws = (char*)d_ws;
  size_t off = 0;
  auto alloc = [&](size_t bytes) -> void* {
    void* p = ws + off;
    off += (bytes + 255) & ~(size_t)255;
    return p;
  };
  float*  x  = (float*) alloc((size_t)B_ * S_ * D_ * 4);
  bf16_t* h  = (bf16_t*)alloc((size_t)B_ * S_ * D_ * 2);
  bf16_t* q  = (bf16_t*)alloc((size_t)B_ * S_ * D_ * 2);
  bf16_t* k  = (bf16_t*)alloc((size_t)B_ * S_ * D_ * 2);
  bf16_t* v  = (bf16_t*)alloc((size_t)B_ * S_ * D_ * 2);
  bf16_t* o  = (bf16_t*)alloc((size_t)B_ * S_ * D_ * 2);
  bf16_t* ff = (bf16_t*)alloc((size_t)B_ * S_ * FF_ * 2);
  bf16_t* xn = (bf16_t*)alloc((size_t)B_ * S_ * D_ * 2);

  const int M = B_ * S_;  // 2180
  auto gemm = [&](const bf16_t* A, const float* W, float* Cf, bf16_t* Cb,
                  const float* resid, int Mm, int Nn, int Kk, int epi, int wBT) {
    dim3 grid((Nn + 63) / 64, (Mm + 127) / 128);
    if (wBT == 0 && epi == 1)
      gemm_wmma_kernel<1, 0><<<grid, 256, 0, stream>>>(A, W, Cf, Cb, resid, Mm, Nn, Kk);
    else if (wBT == 0 && epi == 2)
      gemm_wmma_kernel<2, 0><<<grid, 256, 0, stream>>>(A, W, Cf, Cb, resid, Mm, Nn, Kk);
    else if (wBT == 0 && epi == 3)
      gemm_wmma_kernel<3, 0><<<grid, 256, 0, stream>>>(A, W, Cf, Cb, resid, Mm, Nn, Kk);
    else
      gemm_wmma_kernel<0, 1><<<grid, 256, 0, stream>>>(A, W, Cf, Cb, resid, Mm, Nn, Kk);
  };

  build_x_kernel<<<dim3(S_, B_), 256, 0, stream>>>(tok, mem, embed, x);

  const int qtiles = (S_ + 15) / 16;           // 69
  const int ablocks = (qtiles + 3) / 4;        // 18

  for (int l = 0; l < L_; ++l) {
    const size_t wo = (size_t)l * D_ * D_;
    rmsnorm_kernel<<<M, 256, 0, stream>>>(x, n1 + (size_t)l * D_, h);
    gemm(h, Wq + wo, nullptr, q, nullptr, M, D_, D_, 1, 0);
    gemm(h, Wk + wo, nullptr, k, nullptr, M, D_, D_, 1, 0);
    gemm(h, Wv + wo, nullptr, v, nullptr, M, D_, D_, 1, 0);
    rope_kernel<<<dim3(S_, H_, B_), 32, 0, stream>>>(q, k);
    attention_kernel<<<dim3(ablocks, H_, B_), 128, 0, stream>>>(q, k, v, o);
    gemm(o, Wo + wo, x, nullptr, x, M, D_, D_, 2, 0);
    rmsnorm_kernel<<<M, 256, 0, stream>>>(x, n2 + (size_t)l * D_, h);
    gemm(h, Wup + (size_t)l * D_ * FF_, nullptr, ff, nullptr, M, FF_, D_, 3, 0);
    gemm(ff, Wdn + (size_t)l * FF_ * D_, x, nullptr, x, M, D_, FF_, 2, 0);
  }

  rmsnorm_kernel<<<M, 256, 0, stream>>>(x, fnw, xn);

  for (int b = 0; b < B_; ++b) {
    gemm(xn + ((size_t)b * S_ + N_MEM_) * D_, embed,
         logits + (size_t)b * T_ * V_, nullptr, nullptr, T_, V_, D_, 0, 1);
  }

  halt_kernel<<<dim3(2, B_), 256, 0, stream>>>(xn, hW, hb, halt);
}
